// BLUE_Net_1709396984500
// MI455X (gfx1250) — matrix operand order repaired and checked by reading, not compiled, not run on hardware
//
#include <hip/hip_runtime.h>
#include <hip/hip_bf16.h>
#include <stdint.h>

// ---------------------------------------------------------------------------
// BLUE-Net basic_block for MI455X (gfx1250, wave32, WMMA).
// Heavy convs (64->64 3x3, 128->64 1x1) run as implicit GEMM on
// v_wmma_f32_16x16x32_f16 with f16 NHWC activations, LDS halo tiles for A,
// and weights pre-swizzled into per-lane B fragments (k = 16*(lane/16) + e, same order as A).
// ---------------------------------------------------------------------------

typedef _Float16 half_t;
typedef __attribute__((ext_vector_type(16))) _Float16 v16h;
typedef __attribute__((ext_vector_type(8)))  _Float16 v8h;
typedef __attribute__((ext_vector_type(8)))  float    v8f;

static constexpr int kBS = 8;
static constexpr int kH  = 256;
static constexpr int kW  = 256;
static constexpr int kHW = kH * kW;           // 65536
static constexpr int kC  = 64;

// ---------------- workspace layout (bytes) ----------------
static constexpr size_t WS_MEANS   = 0;                       // 24 f32
static constexpr size_t WS_SORT    = 256;                     // idx_s/m/l (8 int each) + Jlbar (8 f32)
static constexpr size_t WS_BMEAN   = 512;                     // 24 f32
static constexpr size_t WS_RDNIN   = 1024;                    // 8*3*65536 f32 = 6291456 B
static constexpr size_t WS_FRAG33  = WS_RDNIN + 6291456;      // 4 x 73728 B (sfe2, rdb0d, rdb1d, gff2)
static constexpr size_t WS_FRAG11  = WS_FRAG33 + 4*73728;     // 3 x 16384 B (rdb0l, rdb1l, gff1)
static constexpr size_t WS_ACT     = WS_FRAG11 + 3*16384;     // 4 x 67108864 B NHWC f16 activations
static constexpr size_t ACT_BYTES  = (size_t)kBS * kHW * kC * sizeof(half_t);

// ---------------- output layout (f32 elements) ----------------
static constexpr size_t O_B  = 0;
static constexpr size_t O_T  = 1572864;
static constexpr size_t O_J  = 3145728;
static constexpr size_t O_Y  = 4718592;
static constexpr size_t O_Z  = 6291456;
static constexpr size_t O_Q  = 7864320;
static constexpr size_t O_R  = 9437184;
static constexpr size_t O_U  = 11010048;
static constexpr size_t O_V  = 11534336;
static constexpr size_t O_W1 = 12058624;
static constexpr size_t O_W2 = 12582912;
static constexpr size_t O_BETA = 13107200;

// ===========================================================================
// Phase A: means, sort, elementwise updates
// ===========================================================================

__global__ __launch_bounds__(256)
void k_mean_J(const float* __restrict__ J, float* __restrict__ meansJ) {
  __shared__ float red[256];
  const int nc = blockIdx.x;                       // 0..23 = n*3+c
  const float* p = J + (size_t)nc * kHW;
  float s = 0.f;
  for (int i = threadIdx.x; i < kHW; i += 256) s += p[i];
  red[threadIdx.x] = s;
  __syncthreads();
  for (int off = 128; off > 0; off >>= 1) {
    if (threadIdx.x < off) red[threadIdx.x] += red[threadIdx.x + off];
    __syncthreads();
  }
  if (threadIdx.x == 0) meansJ[nc] = red[0] * (1.f / kHW);
}

__global__ void k_sort3(const float* __restrict__ meansJ,
                        int* __restrict__ idx_s, int* __restrict__ idx_m,
                        int* __restrict__ idx_l, float* __restrict__ Jlbar) {
  const int n = threadIdx.x;
  if (n >= kBS) return;
  float m0 = meansJ[n*3+0], m1 = meansJ[n*3+1], m2 = meansJ[n*3+2];
  int i0 = 0, i1 = 1, i2 = 2;
  if (m0 > m1) { float t = m0; m0 = m1; m1 = t; int ti = i0; i0 = i1; i1 = ti; }
  if (m1 > m2) { float t = m1; m1 = m2; m2 = t; int ti = i1; i1 = i2; i2 = ti; }
  if (m0 > m1) { float t = m0; m0 = m1; m1 = t; int ti = i0; i0 = i1; i1 = ti; }
  idx_s[n] = i0; idx_m[n] = i1; idx_l[n] = i2; Jlbar[n] = m2;
}

__global__ __launch_bounds__(256)
void k_phaseA1(const float* __restrict__ J_in, const float* __restrict__ u_in,
               const float* __restrict__ v_in, const float* __restrict__ w1_in,
               const float* __restrict__ w2_in,
               const int* __restrict__ idx_s, const int* __restrict__ idx_m,
               const int* __restrict__ idx_l, const float* __restrict__ Jlbar_a,
               const float* __restrict__ lam1_p, const float* __restrict__ lam2_p,
               float* __restrict__ J_out, float* __restrict__ u_out,
               float* __restrict__ v_out, float* __restrict__ w1_out,
               float* __restrict__ w2_out) {
  const size_t i = (size_t)blockIdx.x * 256 + threadIdx.x;   // 0..524287
  const int n  = (int)(i >> 16);
  const int hw = (int)(i & 65535);
  const float l1 = lam1_p[0], l2 = lam2_p[0];
  const int is = idx_s[n], im = idx_m[n], il = idx_l[n];
  const float Jlb = Jlbar_a[n];
  const size_t base = (size_t)n * 3 * kHW + hw;
  const float Jl = J_in[base + (size_t)il * kHW];
  const float Jm = J_in[base + (size_t)im * kHW];
  const float Js = J_in[base + (size_t)is * kHW];
  const float u = u_in[i], v = v_in[i], w1 = w1_in[i], w2 = w2_in[i];
  const float Jmb = Jl - u + w1 / l1;
  const float Jsb = Jl - v + w2 / l2;
  const float Jm_n = Jm + (Jlb - Jmb) * Jl;
  const float Js_n = Js + (Jlb - Jsb) * Jl;
  const float a1 = Jlb - Jmb + w1 / l1;
  const float a2 = Jlb - Jsb + w2 / l2;
  const float u_n = copysignf(fmaxf(fabsf(a1) - 1.f / l1, 0.f), a1);
  const float v_n = copysignf(fmaxf(fabsf(a2) - 1.f / l2, 0.f), a2);
  J_out[base + (size_t)il * kHW] = Jl;
  J_out[base + (size_t)im * kHW] = Jm_n;
  J_out[base + (size_t)is * kHW] = Js_n;
  u_out[i]  = u_n;
  v_out[i]  = v_n;
  w1_out[i] = w1 + l1 * (Jlb - Jmb - u_n);
  w2_out[i] = w2 + l2 * (Jlb - Jsb - v_n);
}

__global__ __launch_bounds__(256)
void k_Bmean(const float* __restrict__ Jn, const float* __restrict__ t_in,
             const float* __restrict__ I_in, const float* __restrict__ Bp,
             float* __restrict__ Bmean) {
  __shared__ float red[256];
  const int nc = blockIdx.x;
  const size_t base = (size_t)nc * kHW;
  const float g1 = 0.3f;
  float s = 0.f;
  for (int i = threadIdx.x; i < kHW; i += 256) {
    const float t = t_in[base + i], J = Jn[base + i];
    const float I = I_in[base + i], bp = Bp[base + i];
    const float omt = 1.f - t;
    s += (g1 * bp - (J * t - I) * omt) / (omt * omt + g1);
  }
  red[threadIdx.x] = s;
  __syncthreads();
  for (int off = 128; off > 0; off >>= 1) {
    if (threadIdx.x < off) red[threadIdx.x] += red[threadIdx.x + off];
    __syncthreads();
  }
  if (threadIdx.x == 0) Bmean[nc] = red[0] * (1.f / kHW);
}

__global__ __launch_bounds__(256)
void k_phaseA2(const float* __restrict__ I_in, const float* __restrict__ tp_in,
               const float* __restrict__ J_cur, const float* __restrict__ Y_in,
               const float* __restrict__ Z_in, const float* __restrict__ Q_in,
               const float* __restrict__ R_in, const float* __restrict__ Bmean,
               const float* __restrict__ t1dw, const float* __restrict__ beta_p,
               const float* __restrict__ eta_p,
               float* __restrict__ B_out, float* __restrict__ t_out,
               float* __restrict__ J_out, float* __restrict__ Y_out,
               float* __restrict__ Q_out, float* __restrict__ rdn_in) {
  const size_t i = (size_t)blockIdx.x * 256 + threadIdx.x;
  const int n  = (int)(i >> 16);
  const int hw = (int)(i & 65535);
  const float beta = beta_p[0], eta = eta_p[0];
  const float g2 = 0.7f;
  const size_t b0 = (size_t)n * 3 * kHW + hw;
  float s = 0.f;
  #pragma unroll
  for (int c = 0; c < 3; c++) {
    const size_t p = b0 + (size_t)c * kHW;
    const float Bm = Bmean[n*3 + c];
    const float JmB = J_cur[p] - Bm;
    const float num = g2 * tp_in[p] + eta * Z_in[p] - R_in[p] - (Bm - I_in[p]) * JmB;
    const float den = JmB * JmB + g2 + eta;
    s += t1dw[c] * (num / den);
  }
  const float t1 = fmaxf(s, 0.f);
  #pragma unroll
  for (int c = 0; c < 3; c++) {
    const size_t p = b0 + (size_t)c * kHW;
    const float Bm = Bmean[n*3 + c];
    const float I = I_in[p], Y = Y_in[p], Q = Q_in[p];
    const float Jn = (beta * Y - Q - (Bm * (1.f - t1) - I) * t1) / (t1 * t1 + beta);
    B_out[p] = Bm;
    t_out[p] = t1;
    J_out[p] = Jn;
    Y_out[p] = Y;
    Q_out[p] = Q + beta * (Jn - Y);
    rdn_in[p] = t1 + R_in[p] / eta;
  }
}

// ===========================================================================
// Weight pre-swizzle into WMMA B-fragment layout.
// Fragment element (kc, nt, lane, e):
//   K_in_chunk = (lane>=16 ? 16 : 0) + e,  N = nt*16 + (lane&15)
// ===========================================================================

__global__ __launch_bounds__(256)
void k_prep3(const float* __restrict__ w, half_t* __restrict__ frag) {
  const int t = blockIdx.x * 256 + threadIdx.x;     // < 36864
  const int e = t & 15, lane = (t >> 4) & 31, nt = (t >> 9) & 3, kc = t >> 11;
  const int tap = kc >> 1, chunk = kc & 1;
  const int Kin = ((lane >> 4) << 4) + e;
  const int ci = chunk * 32 + Kin;
  const int co = nt * 16 + (lane & 15);
  const int ky = tap / 3, kx = tap - 3 * (tap / 3);
  frag[t] = (half_t)w[(((size_t)co * 64 + ci) * 3 + ky) * 3 + kx];
}

__global__ __launch_bounds__(256)
void k_prep1(const float* __restrict__ w, half_t* __restrict__ frag) {
  const int t = blockIdx.x * 256 + threadIdx.x;     // < 8192
  const int e = t & 15, lane = (t >> 4) & 31, nt = (t >> 9) & 3, kc = t >> 11;
  const int Kin = ((lane >> 4) << 4) + e;
  const int ci = kc * 32 + Kin;
  const int co = nt * 16 + (lane & 15);
  frag[t] = (half_t)w[(size_t)co * 128 + ci];
}

// ===========================================================================
// Boundary convs (small FLOPs -> VALU)
// ===========================================================================

__global__ __launch_bounds__(256)
void k_sfe1(const float* __restrict__ x, const float* __restrict__ w,
            const float* __restrict__ b, half_t* __restrict__ out) {
  __shared__ float wl[64 * 27];
  __shared__ float bl[64];
  for (int i = threadIdx.x; i < 64 * 27; i += 256) wl[i] = w[i];
  if (threadIdx.x < 64) bl[threadIdx.x] = b[threadIdx.x];
  __syncthreads();
  const size_t i = (size_t)blockIdx.x * 256 + threadIdx.x;
  const int n = (int)(i >> 16), hw = (int)(i & 65535);
  const int y = hw >> 8, xp = hw & 255;
  float xv[27];
  int j = 0;
  #pragma unroll
  for (int c = 0; c < 3; c++)
    #pragma unroll
    for (int ky = 0; ky < 3; ky++)
      #pragma unroll
      for (int kx = 0; kx < 3; kx++) {
        const int gy = y + ky - 1, gx = xp + kx - 1;
        xv[j++] = (gy >= 0 && gy < kH && gx >= 0 && gx < kW)
                      ? x[(size_t)(n * 3 + c) * kHW + gy * kW + gx] : 0.f;
      }
  half_t* o = out + ((size_t)n * kHW + hw) * kC;
  for (int co = 0; co < 64; co++) {
    float s = bl[co];
    const float* wr = &wl[co * 27];
    #pragma unroll
    for (int k = 0; k < 27; k++) s += xv[k] * wr[k];
    o[co] = (half_t)s;
  }
}

__global__ __launch_bounds__(256)
void k_outconv(const half_t* __restrict__ g, const float* __restrict__ w,
               const float* __restrict__ b, float* __restrict__ Z) {
  __shared__ float wl[3 * 64 * 9];                  // [co][ci][tap] == OIHW flat
  for (int i = threadIdx.x; i < 1728; i += 256) wl[i] = w[i];
  __syncthreads();
  const size_t i = (size_t)blockIdx.x * 256 + threadIdx.x;
  const int n = (int)(i >> 16), hw = (int)(i & 65535);
  const int y = hw >> 8, x = hw & 255;
  float a0 = b[0], a1 = b[1], a2 = b[2];
  for (int ky = 0; ky < 3; ky++) {
    const int gy = y + ky - 1;
    if (gy < 0 || gy >= kH) continue;
    for (int kx = 0; kx < 3; kx++) {
      const int gx = x + kx - 1;
      if (gx < 0 || gx >= kW) continue;
      const half_t* p = g + ((size_t)n * kHW + gy * kW + gx) * kC;
      const int tap = ky * 3 + kx;
      for (int ci = 0; ci < 64; ci++) {
        const float xv = (float)p[ci];
        a0 += xv * wl[(0 * 64 + ci) * 9 + tap];
        a1 += xv * wl[(1 * 64 + ci) * 9 + tap];
        a2 += xv * wl[(2 * 64 + ci) * 9 + tap];
      }
    }
  }
  Z[(size_t)(n * 3 + 0) * kHW + hw] = a0;
  Z[(size_t)(n * 3 + 1) * kHW + hw] = a1;
  Z[(size_t)(n * 3 + 2) * kHW + hw] = a2;
}

// ===========================================================================
// WMMA 3x3 conv, 64->64, NHWC f16. 8 waves/block, each wave: 16 pixels x
// 64 couts (4 accumulator tiles), K = 9 taps x 2 chunks of 32 channels.
// ===========================================================================

template <bool RELU, bool RES>
__global__ __launch_bounds__(256)
void k_conv3_wmma(const half_t* __restrict__ in, const half_t* __restrict__ frag,
                  const float* __restrict__ bias, const half_t* __restrict__ res,
                  half_t* __restrict__ out) {
  __shared__ half_t sA[3 * 130 * 64];               // 49,920 B halo tile
  const int b = blockIdx.x;
  const int xb = b & 1, y = (b >> 1) & 255, n = b >> 9;
  const int x0 = xb * 128;

  // cooperative halo fill: 390 pixel slots x 128 B
  for (int s = threadIdx.x; s < 390; s += 256) {
    const int r = s / 130, xx = s - r * 130;
    const int gy = y + r - 1, gx = x0 + xx - 1;
    uint4* dst = (uint4*)&sA[s * 64];
    if (gy >= 0 && gy < kH && gx >= 0 && gx < kW) {
      const uint4* src = (const uint4*)(in + ((size_t)n * kHW + gy * kW + gx) * kC);
      #pragma unroll
      for (int q = 0; q < 8; q++) dst[q] = src[q];
    } else {
      const uint4 z = {0u, 0u, 0u, 0u};
      #pragma unroll
      for (int q = 0; q < 8; q++) dst[q] = z;
    }
  }
  __syncthreads();

  const int wave = threadIdx.x >> 5, lane = threadIdx.x & 31;
  const int laneOff = (lane >> 4) * 16;             // A/B fragment lane split
  const int mrow = lane & 15;
  if (lane == 0) __builtin_prefetch(frag, 0, 3);

  v8f acc[4];
  #pragma unroll
  for (int nt = 0; nt < 4; nt++) {
    const float bv = bias[nt * 16 + (lane & 15)];
    #pragma unroll
    for (int r = 0; r < 8; r++) acc[nt][r] = bv;
  }

  #pragma unroll
  for (int tap = 0; tap < 9; tap++) {
    const int ky = tap / 3, kx = tap - 3 * (tap / 3);
    #pragma unroll
    for (int ch = 0; ch < 2; ch++) {
      const int kc = tap * 2 + ch;
      const half_t* ap = &sA[((ky * 130) + (wave * 16 + mrow + kx)) * 64 + ch * 32 + laneOff];
      const v8h lo = *(const v8h*)ap;
      const v8h hi = *(const v8h*)(ap + 8);
      const v16h a = __builtin_shufflevector(lo, hi, 0,1,2,3,4,5,6,7,8,9,10,11,12,13,14,15);
      #pragma unroll
      for (int nt = 0; nt < 4; nt++) {
        const v16h bf = *(const v16h*)(frag + ((size_t)(kc * 4 + nt) * 32 + lane) * 16);
        acc[nt] = __builtin_amdgcn_wmma_f32_16x16x32_f16(false, a, false, bf,
                                                         (short)0, acc[nt], false, false);
      }
    }
  }

  const int pxbase = x0 + wave * 16;
  #pragma unroll
  for (int nt = 0; nt < 4; nt++) {
    const int co = nt * 16 + (lane & 15);
    #pragma unroll
    for (int r = 0; r < 8; r++) {
      const int px = pxbase + r + ((lane >> 4) << 3);
      const size_t oi = ((size_t)n * kHW + y * kW + px) * kC + co;
      float v = acc[nt][r];
      if (RELU) v = fmaxf(v, 0.f);
      if (RES)  v += (float)res[oi];
      out[oi] = (half_t)v;
    }
  }
}

// ===========================================================================
// WMMA 1x1 conv, concat(64,64)->64, NHWC f16. K = 4 chunks of 32 (2 per input).
// ===========================================================================

template <bool RES>
__global__ __launch_bounds__(256)
void k_conv1_wmma(const half_t* __restrict__ inA, const half_t* __restrict__ inB,
                  const half_t* __restrict__ frag, const float* __restrict__ bias,
                  const half_t* __restrict__ res, half_t* __restrict__ out) {
  const int wave = threadIdx.x >> 5, lane = threadIdx.x & 31;
  const int laneOff = (lane >> 4) * 16;
  const size_t pix0 = (size_t)blockIdx.x * 128 + wave * 16;
  const size_t prow = pix0 + (lane & 15);

  v8f acc[4];
  #pragma unroll
  for (int nt = 0; nt < 4; nt++) {
    const float bv = bias[nt * 16 + (lane & 15)];
    #pragma unroll
    for (int r = 0; r < 8; r++) acc[nt][r] = bv;
  }

  #pragma unroll
  for (int kc = 0; kc < 4; kc++) {
    const half_t* src = (kc < 2) ? inA : inB;
    const int ciOff = (kc & 1) * 32;
    const half_t* ap = src + prow * kC + ciOff + laneOff;
    const v8h lo = *(const v8h*)ap;
    const v8h hi = *(const v8h*)(ap + 8);
    const v16h a = __builtin_shufflevector(lo, hi, 0,1,2,3,4,5,6,7,8,9,10,11,12,13,14,15);
    #pragma unroll
    for (int nt = 0; nt < 4; nt++) {
      const v16h bf = *(const v16h*)(frag + ((size_t)(kc * 4 + nt) * 32 + lane) * 16);
      acc[nt] = __builtin_amdgcn_wmma_f32_16x16x32_f16(false, a, false, bf,
                                                       (short)0, acc[nt], false, false);
    }
  }

  #pragma unroll
  for (int nt = 0; nt < 4; nt++) {
    const int co = nt * 16 + (lane & 15);
    #pragma unroll
    for (int r = 0; r < 8; r++) {
      const size_t pix = pix0 + r + ((lane >> 4) << 3);
      const size_t oi = pix * kC + co;
      float v = acc[nt][r];
      if (RES) v += (float)res[oi];
      out[oi] = (half_t)v;
    }
  }
}

// ===========================================================================
// Phase C: R update + beta passthrough
// ===========================================================================

__global__ __launch_bounds__(256)
void k_phaseC(const float* __restrict__ R_in, const float* __restrict__ t_out,
              const float* __restrict__ Z_out, const float* __restrict__ eta_p,
              const float* __restrict__ beta_p, float* __restrict__ R_out,
              float* __restrict__ beta_out) {
  const size_t i = (size_t)blockIdx.x * 256 + threadIdx.x;
  if (i < (size_t)kBS * 3 * kHW) {
    const float eta = eta_p[0];
    R_out[i] = R_in[i] + eta * (t_out[i] - Z_out[i]);
  }
  if (i == 0) beta_out[0] = beta_p[0];
}

// ===========================================================================
// Launch
// ===========================================================================

extern "C" void kernel_launch(void* const* d_in, const int* in_sizes, int n_in,
                              void* d_out, int out_size, void* d_ws, size_t ws_size,
                              hipStream_t stream) {
  // inputs (setup_inputs dict order; rdn_params flattened in insertion order)
  const float* I    = (const float*)d_in[0];
  const float* t_p  = (const float*)d_in[1];
  const float* B_p  = (const float*)d_in[2];
  // d_in[3] = B (unused: recomputed), d_in[4] = t
  const float* t_in = (const float*)d_in[4];
  const float* J_in = (const float*)d_in[5];
  const float* Y_in = (const float*)d_in[6];
  const float* Z_in = (const float*)d_in[7];
  const float* Q_in = (const float*)d_in[8];
  const float* R_in = (const float*)d_in[9];
  const float* u_in = (const float*)d_in[10];
  const float* v_in = (const float*)d_in[11];
  const float* w1_in = (const float*)d_in[12];
  const float* w2_in = (const float*)d_in[13];
  const float* sfe1_w = (const float*)d_in[14];
  const float* sfe1_b = (const float*)d_in[15];
  const float* sfe2_w = (const float*)d_in[16];
  const float* sfe2_b = (const float*)d_in[17];
  const float* rdb0_dw = (const float*)d_in[18];
  const float* rdb0_db = (const float*)d_in[19];
  const float* rdb0_lw = (const float*)d_in[20];
  const float* rdb0_lb = (const float*)d_in[21];
  const float* rdb1_dw = (const float*)d_in[22];
  const float* rdb1_db = (const float*)d_in[23];
  const float* rdb1_lw = (const float*)d_in[24];
  const float* rdb1_lb = (const float*)d_in[25];
  const float* gff1_w = (const float*)d_in[26];
  const float* gff1_b = (const float*)d_in[27];
  const float* gff2_w = (const float*)d_in[28];
  const float* gff2_b = (const float*)d_in[29];
  const float* out_w = (const float*)d_in[30];
  const float* out_b = (const float*)d_in[31];
  const float* t1d_w = (const float*)d_in[32];
  const float* beta_p = (const float*)d_in[33];
  const float* eta_p  = (const float*)d_in[34];
  const float* lam1_p = (const float*)d_in[35];
  const float* lam2_p = (const float*)d_in[36];

  float* out = (float*)d_out;
  char* ws = (char*)d_ws;

  float* meansJ = (float*)(ws + WS_MEANS);
  int*   idx_s  = (int*)(ws + WS_SORT);
  int*   idx_m  = idx_s + 8;
  int*   idx_l  = idx_m + 8;
  float* Jlbar  = (float*)(idx_l + 8);
  float* Bmean  = (float*)(ws + WS_BMEAN);
  float* rdn_in = (float*)(ws + WS_RDNIN);

  half_t* frag_sfe2  = (half_t*)(ws + WS_FRAG33 + 0 * 73728);
  half_t* frag_rdb0d = (half_t*)(ws + WS_FRAG33 + 1 * 73728);
  half_t* frag_rdb1d = (half_t*)(ws + WS_FRAG33 + 2 * 73728);
  half_t* frag_gff2  = (half_t*)(ws + WS_FRAG33 + 3 * 73728);
  half_t* frag_rdb0l = (half_t*)(ws + WS_FRAG11 + 0 * 16384);
  half_t* frag_rdb1l = (half_t*)(ws + WS_FRAG11 + 1 * 16384);
  half_t* frag_gff1  = (half_t*)(ws + WS_FRAG11 + 2 * 16384);

  half_t* actA = (half_t*)(ws + WS_ACT + 0 * ACT_BYTES);   // sfe1 (kept for gff2 residual)
  half_t* actB = (half_t*)(ws + WS_ACT + 1 * ACT_BYTES);
  half_t* actC = (half_t*)(ws + WS_ACT + 2 * ACT_BYTES);
  half_t* actD = (half_t*)(ws + WS_ACT + 3 * ACT_BYTES);

  const dim3 blk(256);
  const int gridPix  = (kBS * kHW) / 256;          // 2048
  const int gridConv = 4096;                        // 8 imgs * 256 rows * 2 xblocks
  const int gridC    = (kBS * 3 * kHW + 255) / 256; // 6144

  // weight pre-swizzle (independent of phase A)
  k_prep3<<<144, blk, 0, stream>>>(sfe2_w,  frag_sfe2);
  k_prep3<<<144, blk, 0, stream>>>(rdb0_dw, frag_rdb0d);
  k_prep3<<<144, blk, 0, stream>>>(rdb1_dw, frag_rdb1d);
  k_prep3<<<144, blk, 0, stream>>>(gff2_w,  frag_gff2);
  k_prep1<<<32,  blk, 0, stream>>>(rdb0_lw, frag_rdb0l);
  k_prep1<<<32,  blk, 0, stream>>>(rdb1_lw, frag_rdb1l);
  k_prep1<<<32,  blk, 0, stream>>>(gff1_w,  frag_gff1);

  // phase A
  k_mean_J<<<24, blk, 0, stream>>>(J_in, meansJ);
  k_sort3<<<1, 32, 0, stream>>>(meansJ, idx_s, idx_m, idx_l, Jlbar);
  k_phaseA1<<<gridPix, blk, 0, stream>>>(J_in, u_in, v_in, w1_in, w2_in,
                                         idx_s, idx_m, idx_l, Jlbar, lam1_p, lam2_p,
                                         out + O_J, out + O_U, out + O_V,
                                         out + O_W1, out + O_W2);
  k_Bmean<<<24, blk, 0, stream>>>(out + O_J, t_in, I, B_p, Bmean);
  k_phaseA2<<<gridPix, blk, 0, stream>>>(I, t_p, out + O_J, Y_in, Z_in, Q_in, R_in,
                                         Bmean, t1d_w, beta_p, eta_p,
                                         out + O_B, out + O_T, out + O_J, out + O_Y,
                                         out + O_Q, rdn_in);

  // RDN forward (WMMA pipeline)
  k_sfe1<<<gridPix, blk, 0, stream>>>(rdn_in, sfe1_w, sfe1_b, actA);                 // sfe1
  k_conv3_wmma<false, false><<<gridConv, blk, 0, stream>>>(actA, frag_sfe2, sfe2_b,
                                                           nullptr, actB);           // h0
  k_conv3_wmma<true,  false><<<gridConv, blk, 0, stream>>>(actB, frag_rdb0d, rdb0_db,
                                                           nullptr, actC);           // r0
  k_conv1_wmma<true><<<gridConv, blk, 0, stream>>>(actB, actC, frag_rdb0l, rdb0_lb,
                                                   actB, actD);                      // h1 = h0 + lff
  k_conv3_wmma<true,  false><<<gridConv, blk, 0, stream>>>(actD, frag_rdb1d, rdb1_db,
                                                           nullptr, actB);           // r1
  k_conv1_wmma<true><<<gridConv, blk, 0, stream>>>(actD, actB, frag_rdb1l, rdb1_lb,
                                                   actD, actC);                      // h2 = h1 + lff
  k_conv1_wmma<false><<<gridConv, blk, 0, stream>>>(actD, actC, frag_gff1, gff1_b,
                                                    nullptr, actB);                  // g0
  k_conv3_wmma<false, true><<<gridConv, blk, 0, stream>>>(actB, frag_gff2, gff2_b,
                                                          actA, actD);               // g1 = gff2 + sfe1
  k_outconv<<<gridPix, blk, 0, stream>>>(actD, out_w, out_b, out + O_Z);             // Z

  // phase C
  k_phaseC<<<gridC, blk, 0, stream>>>(R_in, out + O_T, out + O_Z, eta_p, beta_p,
                                      out + O_R, out + O_BETA);
}
